// GCN_23313082483287
// MI455X (gfx1250) — matrix operand order repaired and checked by reading, not compiled, not run on hardware
//
#include <hip/hip_runtime.h>
#include <hip/hip_bf16.h>

typedef __attribute__((ext_vector_type(16))) _Float16 v16h;
typedef __attribute__((ext_vector_type(8)))  _Float16 v8h;
typedef __attribute__((ext_vector_type(8)))  float    v8f;

static constexpr int DL   = 128;      // latent dim
static constexpr int DIN  = 64;       // input feature dim
static constexpr int ROWS = 64;       // node rows per block
static constexpr int TPB  = 128;      // 4 waves of 32
static constexpr int SA   = DL + 8;   // LDS activation stride (halves): 16B-aligned, bank-skewed
static constexpr int SW   = DL + 8;
static constexpr int SAE  = DIN + 8;

// ---- WMMA fragment loaders (wave32, contiguous k = 16*half + s, same order for A and B) ----
// A 16x32 f16: lane(0-15)=row M, wave half holds K = half*16..+15
__device__ __forceinline__ v16h ld_frag_a(const _Float16* p) {
  v8h lo = *(const v8h*)p;
  v8h hi = *(const v8h*)(p + 8);
  return __builtin_shufflevector(lo, hi, 0,1,2,3,4,5,6,7,8,9,10,11,12,13,14,15);
}
// B 32x16 f16 staged as W^T rows: lane(0-15)=col N, wave half holds K = half*16..+15
__device__ __forceinline__ v16h ld_frag_b(const _Float16* p) {
  v8h lo = *(const v8h*)p;
  v8h hi = *(const v8h*)(p + 8);
  return __builtin_shufflevector(lo, hi, 0,1,2,3,4,5,6,7,8,9,10,11,12,13,14,15);
}

__device__ __forceinline__ void atomic_add_f32(float* p, float v) {
  unsafeAtomicAdd(p, v);            // native global_atomic_add_f32
}

// ---------------- embed: h = node_feats[N x 64] @ W_embed[64 x 128] + b ----------------
__global__ __launch_bounds__(TPB)
void embed_kernel(const float* __restrict__ nf, const float* __restrict__ We,
                  const float* __restrict__ be, float* __restrict__ out, int nrows)
{
  __shared__ __attribute__((aligned(16))) _Float16 act[ROWS * SAE];
  __shared__ __attribute__((aligned(16))) _Float16 wt[DL * SAE];
  __shared__ float bs[DL];

  const int tid  = threadIdx.x;
  const int lane = tid & 31, wave = tid >> 5;
  const int ln16 = lane & 15, half = lane >> 4;
  const int rb   = blockIdx.x * ROWS;
  const bool full = (rb + ROWS) <= nrows;          // uniform fast path (all but tail block)

  if (full) {
    for (int idx = tid; idx < ROWS * DIN; idx += TPB) {
      int r = idx >> 6, c = idx & (DIN - 1);
      act[r * SAE + c] = (_Float16)nf[(size_t)(rb + r) * DIN + c];
    }
  } else {
    for (int idx = tid; idx < ROWS * DIN; idx += TPB) {
      int r = idx >> 6, c = idx & (DIN - 1);
      int gr = rb + r;
      act[r * SAE + c] = (_Float16)(gr < nrows ? nf[(size_t)gr * DIN + c] : 0.f);
    }
  }
  for (int idx = tid; idx < DIN * DL; idx += TPB) {
    int k = idx >> 7, n = idx & (DL - 1);
    wt[n * SAE + k] = (_Float16)We[idx];           // stage W^T as f16
  }
  if (tid < DL) bs[tid] = be[tid];
  __syncthreads();

  const int waveRow = wave * 16;
  v16h afr[2];
#pragma unroll
  for (int kc = 0; kc < 2; ++kc)
    afr[kc] = ld_frag_a(&act[(waveRow + ln16) * SAE + kc * 32 + half * 16]);

  v8f acc[8];
#pragma unroll
  for (int nt = 0; nt < 8; ++nt) {
    v8f c = {};
#pragma unroll
    for (int kc = 0; kc < 2; ++kc) {
      v16h bfr = ld_frag_b(&wt[(nt * 16 + ln16) * SAE + kc * 32 + half * 16]);
      c = __builtin_amdgcn_wmma_f32_16x16x32_f16(false, afr[kc], false, bfr,
                                                 (short)0, c, false, false);
    }
    acc[nt] = c;
  }

  // epilogue: bias add, row-major coalesced stores; row guard only on tail block
#pragma unroll
  for (int v = 0; v < 8; ++v) {
    int gr = rb + waveRow + v + half * 8;
    if (full || gr < nrows) {
#pragma unroll
      for (int nt = 0; nt < 8; ++nt) {
        int col = nt * 16 + ln16;
        out[(size_t)gr * DL + col] = acc[nt][v] + bs[col];
      }
    }
  }
}

// -------- fused 3-layer MLP (Dense+relu+LN, Dense+relu+LN, Dense+relu) * s_norm --------
__global__ __launch_bounds__(TPB)
void mlp_kernel(const float* __restrict__ hin, const float* __restrict__ row_scale,
                const float* __restrict__ W1, const float* __restrict__ b1,
                const float* __restrict__ g1, const float* __restrict__ be1,
                const float* __restrict__ W2, const float* __restrict__ b2,
                const float* __restrict__ g2, const float* __restrict__ be2,
                const float* __restrict__ W3, const float* __restrict__ b3,
                const float* __restrict__ snorm, float* __restrict__ mout, int nrows)
{
  __shared__ __attribute__((aligned(16))) _Float16 act[ROWS * SA];   // 17.4 KB
  __shared__ __attribute__((aligned(16))) _Float16 wt[DL * SW];      // 34.8 KB
  __shared__ float bs[DL], gs[DL], bes[DL];

  const int tid  = threadIdx.x;
  const int lane = tid & 31, wave = tid >> 5;
  const int ln16 = lane & 15, half = lane >> 4;
  const int rb   = blockIdx.x * ROWS;
  const int waveRow = wave * 16;
  const bool full = (rb + ROWS) <= nrows;

  // stage input tile as f16 (optionally fused r_norm row scale); branches hoisted
  if (full) {
    if (row_scale) {
      for (int idx = tid; idx < ROWS * DL; idx += TPB) {
        int r = idx >> 7, c = idx & (DL - 1);
        int gr = rb + r;
        act[r * SA + c] = (_Float16)(hin[(size_t)gr * DL + c] * row_scale[gr]);
      }
    } else {
      for (int idx = tid; idx < ROWS * DL; idx += TPB) {
        int r = idx >> 7, c = idx & (DL - 1);
        act[r * SA + c] = (_Float16)hin[(size_t)(rb + r) * DL + c];
      }
    }
  } else {
    for (int idx = tid; idx < ROWS * DL; idx += TPB) {
      int r = idx >> 7, c = idx & (DL - 1);
      int gr = rb + r;
      float v = 0.f;
      if (gr < nrows) {
        v = hin[(size_t)gr * DL + c];
        if (row_scale) v *= row_scale[gr];
      }
      act[r * SA + c] = (_Float16)v;
    }
  }

  const float* Ws[3]  = {W1, W2, W3};
  const float* Bs[3]  = {b1, b2, b3};
  const float* Gs[2]  = {g1, g2};
  const float* BEs[2] = {be1, be2};

  for (int layer = 0; layer < 3; ++layer) {
    __syncthreads();                               // all waves done with previous wt
    const float* W = Ws[layer];
    for (int idx = tid; idx < DL * DL; idx += TPB) {
      int k = idx >> 7, n = idx & (DL - 1);
      wt[n * SW + k] = (_Float16)W[idx];           // stage W^T as f16
    }
    if (tid < DL) {
      bs[tid] = Bs[layer][tid];
      if (layer < 2) { gs[tid] = Gs[layer][tid]; bes[tid] = BEs[layer][tid]; }
    }
    __syncthreads();

    v16h afr[4];
#pragma unroll
    for (int kc = 0; kc < 4; ++kc)
      afr[kc] = ld_frag_a(&act[(waveRow + ln16) * SA + kc * 32 + half * 16]);

    v8f acc[8];
#pragma unroll
    for (int nt = 0; nt < 8; ++nt) {
      v8f c = {};
#pragma unroll
      for (int kc = 0; kc < 4; ++kc) {
        v16h bfr = ld_frag_b(&wt[(nt * 16 + ln16) * SW + kc * 32 + half * 16]);
        c = __builtin_amdgcn_wmma_f32_16x16x32_f16(false, afr[kc], false, bfr,
                                                   (short)0, c, false, false);
      }
      acc[nt] = c;
    }

    // bias + relu in f32 on accumulators
#pragma unroll
    for (int nt = 0; nt < 8; ++nt) {
      float b = bs[nt * 16 + ln16];
#pragma unroll
      for (int v = 0; v < 8; ++v) {
        float x = acc[nt][v] + b;
        acc[nt][v] = x > 0.f ? x : 0.f;
      }
    }

    if (layer < 2) {
      // per-row LayerNorm: row M = v + 8*half lives in 16 lanes of one half-wave,
      // 8 partials per lane -> 4 shfl_xor steps (masks<16 stay inside the half-wave)
#pragma unroll
      for (int v = 0; v < 8; ++v) {
        float s1 = 0.f, s2 = 0.f;
#pragma unroll
        for (int nt = 0; nt < 8; ++nt) { float x = acc[nt][v]; s1 += x; s2 += x * x; }
#pragma unroll
        for (int m = 1; m <= 8; m <<= 1) {
          s1 += __shfl_xor(s1, m, 32);
          s2 += __shfl_xor(s2, m, 32);
        }
        float mean = s1 * (1.f / 128.f);
        float var  = s2 * (1.f / 128.f) - mean * mean;
        float inv  = rsqrtf(var + 1e-6f);
        int r = waveRow + v + half * 8;
#pragma unroll
        for (int nt = 0; nt < 8; ++nt) {
          int col = nt * 16 + ln16;
          float y = (acc[nt][v] - mean) * inv * gs[col] + bes[col];
          act[r * SA + col] = (_Float16)y;         // wave-private rows, no barrier needed
        }
      }
    } else {
      // final layer: scale by s_norm and emit f32 message matrix (guard only on tail block)
#pragma unroll
      for (int v = 0; v < 8; ++v) {
        int gr = rb + waveRow + v + half * 8;
        if (full || gr < nrows) {
          float sn = snorm[gr];
#pragma unroll
          for (int nt = 0; nt < 8; ++nt)
            mout[(size_t)gr * DL + nt * 16 + ln16] = acc[nt][v] * sn;
        }
      }
    }
  }
}

// ---------------- graph plumbing kernels ----------------
__global__ void deg_init_kernel(float* sdeg, float* rdeg, int n) {
  int i = blockIdx.x * blockDim.x + threadIdx.x;
  if (i < n) { sdeg[i] = 1.f; rdeg[i] = 1.f; }    // self edges
}
__global__ void deg_edge_kernel(const int* __restrict__ se, const int* __restrict__ re,
                                float* sdeg, float* rdeg, int e) {
  int i = blockIdx.x * blockDim.x + threadIdx.x;
  if (i < e) { atomic_add_f32(&sdeg[se[i]], 1.f); atomic_add_f32(&rdeg[re[i]], 1.f); }
}
__global__ void norm_kernel(const float* sdeg, const float* rdeg,
                            float* sn, float* rn, int n) {
  int i = blockIdx.x * blockDim.x + threadIdx.x;
  if (i < n) { sn[i] = rsqrtf(fmaxf(sdeg[i], 1.f)); rn[i] = rsqrtf(fmaxf(rdeg[i], 1.f)); }
}
__global__ void copy4_kernel(float* __restrict__ dst, const float* __restrict__ src, int n4) {
  int i = blockIdx.x * blockDim.x + threadIdx.x;
  if (i < n4) ((float4*)dst)[i] = ((const float4*)src)[i];   // self-edge contribution = init
}
__global__ void scatter_kernel(const float* __restrict__ m, const int* __restrict__ se,
                               const int* __restrict__ re, float* __restrict__ out, int e) {
  int t = blockIdx.x * blockDim.x + threadIdx.x;
  int ei = t >> 5, c = (t & 31) * 4;
  if (ei >= e) return;
  int s = se[ei], r = re[ei];
  float4 v = *(const float4*)(m + (size_t)s * DL + c);
  float* p = out + (size_t)r * DL + c;
  atomic_add_f32(p + 0, v.x); atomic_add_f32(p + 1, v.y);
  atomic_add_f32(p + 2, v.z); atomic_add_f32(p + 3, v.w);
}
__global__ void zero_kernel(float* p, int n) {
  int i = blockIdx.x * blockDim.x + threadIdx.x;
  if (i < n) p[i] = 0.f;
}
__global__ void count_kernel(const int* __restrict__ gid, float* counts, int n) {
  int i = blockIdx.x * blockDim.x + threadIdx.x;
  if (i < n) atomic_add_f32(&counts[gid[i]], 1.f);
}
__global__ void pool_kernel(const float* __restrict__ h, const float* __restrict__ rn,
                            const int* __restrict__ gid, float* __restrict__ pooled, int n) {
  int t = blockIdx.x * blockDim.x + threadIdx.x;
  int i = t >> 5, c = (t & 31) * 4;
  if (i >= n) return;
  int g = gid[i];
  float s = rn[i];                                  // fused final r_norm
  float4 v = *(const float4*)(h + (size_t)i * DL + c);
  float* p = pooled + (size_t)g * DL + c;
  atomic_add_f32(p + 0, v.x * s); atomic_add_f32(p + 1, v.y * s);
  atomic_add_f32(p + 2, v.z * s); atomic_add_f32(p + 3, v.w * s);
}
__global__ void decoder_kernel(const float* __restrict__ pooled, const float* __restrict__ counts,
                               const float* __restrict__ Wd, const float* __restrict__ bd,
                               float* __restrict__ out, int g_total) {
  int t = blockIdx.x * blockDim.x + threadIdx.x;
  if (t >= g_total * 8) return;
  int g = t >> 3, o = t & 7;
  float inv = 1.f / fmaxf(counts[g], 1.f);
  float s = 0.f;
  for (int k = 0; k < DL; ++k) s += pooled[(size_t)g * DL + k] * Wd[k * 8 + o];
  out[t] = s * inv + bd[o];
}

extern "C" void kernel_launch(void* const* d_in, const int* in_sizes, int n_in,
                              void* d_out, int out_size, void* d_ws, size_t ws_size,
                              hipStream_t stream) {
  (void)n_in; (void)ws_size;
  const float* node_feats = (const float*)d_in[0];
  const int*   senders    = (const int*)d_in[1];
  const int*   receivers  = (const int*)d_in[2];
  const int*   gids       = (const int*)d_in[3];
  const float* W_embed    = (const float*)d_in[4];
  const float* b_embed    = (const float*)d_in[5];
  const float* W1 = (const float*)d_in[6],  *b1 = (const float*)d_in[7];
  const float* g1 = (const float*)d_in[8],  *be1 = (const float*)d_in[9];
  const float* W2 = (const float*)d_in[10], *b2 = (const float*)d_in[11];
  const float* g2 = (const float*)d_in[12], *be2 = (const float*)d_in[13];
  const float* W3 = (const float*)d_in[14], *b3 = (const float*)d_in[15];
  const float* Wd = (const float*)d_in[16], *bd = (const float*)d_in[17];
  float* out = (float*)d_out;

  const int N = in_sizes[0] / DIN;
  const int E = in_sizes[1];
  const int G = out_size / 8;

  char* w = (char*)d_ws;
  auto carve = [&](size_t bytes) { void* p = (void*)w; w += (bytes + 255) & ~(size_t)255; return p; };
  float* hA     = (float*)carve((size_t)N * DL * 4);
  float* hB     = (float*)carve((size_t)N * DL * 4);
  float* mbuf   = (float*)carve((size_t)N * DL * 4);
  float* sdeg   = (float*)carve((size_t)N * 4);
  float* rdeg   = (float*)carve((size_t)N * 4);
  float* snorm  = (float*)carve((size_t)N * 4);
  float* rnorm  = (float*)carve((size_t)N * 4);
  float* pooled = (float*)carve((size_t)G * DL * 4);
  float* counts = (float*)carve((size_t)G * 4);

  const int nblk  = (N + 255) / 256;
  const int eblk  = (E + 255) / 256;
  const int gemmb = (N + ROWS - 1) / ROWS;

  // degrees -> norms
  deg_init_kernel<<<nblk, 256, 0, stream>>>(sdeg, rdeg, N);
  deg_edge_kernel<<<eblk, 256, 0, stream>>>(senders, receivers, sdeg, rdeg, E);
  norm_kernel<<<nblk, 256, 0, stream>>>(sdeg, rdeg, snorm, rnorm, N);

  // embed
  embed_kernel<<<gemmb, TPB, 0, stream>>>(node_feats, W_embed, b_embed, hA, N);

  // 3 message-passing steps
  for (int s = 0; s < 3; ++s) {
    const int off = s * DL * DL, voff = s * DL;
    mlp_kernel<<<gemmb, TPB, 0, stream>>>(hA, s ? rnorm : nullptr,
        W1 + off, b1 + voff, g1 + voff, be1 + voff,
        W2 + off, b2 + voff, g2 + voff, be2 + voff,
        W3 + off, b3 + voff, snorm, mbuf, N);
    copy4_kernel<<<(N * DL / 4 + 255) / 256, 256, 0, stream>>>(hB, mbuf, N * DL / 4);
    scatter_kernel<<<((size_t)E * 32 + 255) / 256, 256, 0, stream>>>(mbuf, senders, receivers, hB, E);
    float* t = hA; hA = hB; hB = t;
  }

  // pool (fused final r_norm) + decode
  zero_kernel<<<(G * (DL + 1) + 255) / 256, 256, 0, stream>>>(pooled, G * DL);
  zero_kernel<<<(G + 255) / 256, 256, 0, stream>>>(counts, G);
  count_kernel<<<nblk, 256, 0, stream>>>(gids, counts, N);
  pool_kernel<<<((size_t)N * 32 + 255) / 256, 256, 0, stream>>>(hA, rnorm, gids, pooled, N);
  decoder_kernel<<<(G * 8 + 255) / 256, 256, 0, stream>>>(pooled, counts, Wd, bd, out, G);
}
